// MultiHeadAttention_35725537968369
// MI455X (gfx1250) — matrix operand order repaired and checked by reading, not compiled, not run on hardware
//
#include <hip/hip_runtime.h>

// ---------------------------------------------------------------------------
// Types for CDNA5 WMMA (wave32): D(16x16 f32) = A(16x32 bf16) x B(32x16 bf16) + C
// ---------------------------------------------------------------------------
typedef __attribute__((ext_vector_type(16))) __bf16 v16bf;
typedef __attribute__((ext_vector_type(8)))  __bf16 v8bf;
typedef __attribute__((ext_vector_type(4)))  __bf16 v4bf;
typedef __attribute__((ext_vector_type(8)))  float  v8f;

// Payload type for async global->LDS b128 (from compiler diagnostic: v4i*)
typedef int v4i_vs __attribute__((vector_size(4 * sizeof(int))));

#define WMMA_BF16(a, b, c) \
  __builtin_amdgcn_wmma_f32_16x16x32_bf16(false, (a), false, (b), (short)0, (c), false, false)

static __device__ __forceinline__ v8f vzero8f() {
  v8f z = {0.f, 0.f, 0.f, 0.f, 0.f, 0.f, 0.f, 0.f};
  return z;
}

// Problem constants
#define kB   2
#define kS   2048
#define kE   768
#define kH   12
#define kD   64
#define kE3  2304
#define kBS  (kB * kS)          // 4096 rows
#define kKT  (kE / 32)          // 24 k-steps

// LDS B-tile: 64 cols x 32 k (bf16), row stride padded to 80B (40 elems) so the
// 16-lane column reads map to 16 distinct banks (80*l mod 256 all distinct).
#define BT_STRIDE 40            // elements per LDS row (64B data + 16B pad)
#define BT_ELEMS  (64 * BT_STRIDE)

union Afrag { v16bf v; v8bf h[2]; };

// --------------------------- async global->LDS -----------------------------
#if __has_builtin(__builtin_amdgcn_global_load_async_to_lds_b128)
#define HAVE_ASYNC_LDS 1
#else
#define HAVE_ASYNC_LDS 0
#endif

static __device__ __forceinline__ void
copy16B_to_lds(const __bf16* gsrc, __bf16* lds_dst) {
#if HAVE_ASYNC_LDS
  __builtin_amdgcn_global_load_async_to_lds_b128(
      (__attribute__((address_space(1))) v4i_vs*)gsrc,
      (__attribute__((address_space(3))) v4i_vs*)lds_dst, 0, 0);
#else
  *(v8bf*)lds_dst = *(const v8bf*)gsrc;
#endif
}

// Wait until at most N async copies remain in flight (per-thread, in order).
static __device__ __forceinline__ void wait_async_le0() {
#if HAVE_ASYNC_LDS
#if __has_builtin(__builtin_amdgcn_s_wait_asynccnt)
  __builtin_amdgcn_s_wait_asynccnt(0);
#else
  asm volatile("s_wait_asynccnt 0x0" ::: "memory");
#endif
#endif
}
static __device__ __forceinline__ void wait_async_le1() {
#if HAVE_ASYNC_LDS
#if __has_builtin(__builtin_amdgcn_s_wait_asynccnt)
  __builtin_amdgcn_s_wait_asynccnt(1);
#else
  asm volatile("s_wait_asynccnt 0x1" ::: "memory");
#endif
#endif
}

// ---------------------------------------------------------------------------
// Pre-pass 1: fp32 -> bf16 elementwise (x), 4 elems/thread
// ---------------------------------------------------------------------------
__global__ __launch_bounds__(256)
void cvt_f32_bf16(const float* __restrict__ in, __bf16* __restrict__ outb, int n4) {
  int idx = blockIdx.x * blockDim.x + threadIdx.x;
  if (idx < n4) {
    float4 v = ((const float4*)in)[idx];
    v4bf o;
    o[0] = (__bf16)v.x; o[1] = (__bf16)v.y; o[2] = (__bf16)v.z; o[3] = (__bf16)v.w;
    ((v4bf*)outb)[idx] = o;
  }
}

// ---------------------------------------------------------------------------
// Pre-pass 2: fp32 W[K][N] -> bf16 W^T[N][K]  (B-matrix columns contiguous)
// ---------------------------------------------------------------------------
__global__ __launch_bounds__(256)
void cvt_w_T(const float* __restrict__ w, __bf16* __restrict__ wt, int K, int N) {
  int idx = blockIdx.x * blockDim.x + threadIdx.x;
  if (idx < K * N) {
    int n = idx / K, k = idx - n * K;
    wt[idx] = (__bf16)w[k * N + n];      // wt[n][k] = w[k][n]
  }
}

// ---------------------------------------------------------------------------
// Shared GEMM core: block computes 128(M) x 64(N).
// B-tile double-buffered in LDS via async global->LDS copies: tile t+1's DMA
// flies while tile t's WMMAs execute (wait asynccnt<=1 exploits in-order
// async completion). Per-wave: 16(M) x 64(N), acc[4] of 16x16 f32.
// ---------------------------------------------------------------------------
static __device__ __forceinline__ void
gemm_core(const __bf16* __restrict__ aptr,   // lane's A row base (xb + row*kE)
          const __bf16* __restrict__ wt,     // W^T [N][kE] bf16
          int n0, int tid, int lo, int hi, __bf16* btile, v8f acc[4]) {
  const int akb = hi * 16;     // A K-base (contiguous k order, matches B)
  const int bkb = hi * 16;     // B K-base (contiguous k order, matches A)
  const int frow = tid >> 2;   // 0..63  : fill row   (one B column)
  const int fchk = tid & 3;    // 0..3   : fill chunk (16B of the 64B row)
  const __bf16* gfill = wt + (n0 + frow) * kE + fchk * 8;
  const int     lfill = frow * BT_STRIDE + fchk * 8;

  // prologue: stage tile 0 into buffer 0
  copy16B_to_lds(gfill, btile + lfill);

  for (int t = 0; t < kKT; ++t) {
    const int k0 = t * 32;
    __bf16* cur = btile + (t & 1) * BT_ELEMS;
    __bf16* nxt = btile + ((t + 1) & 1) * BT_ELEMS;

    __syncthreads();                       // readers of `nxt` (tile t-1) done
    if (t + 1 < kKT) {
      copy16B_to_lds(gfill + k0 + 32, nxt + lfill);  // overlap DMA of t+1
      wait_async_le1();                    // tile t landed; t+1 still in flight
    } else {
      wait_async_le0();                    // last tile fully landed
    }
    __syncthreads();                       // tile t visible to all waves

    __builtin_prefetch(aptr + k0 + 64, 0, 0);
    Afrag a;
    a.h[0] = *(const v8bf*)(aptr + k0 + akb);
    a.h[1] = *(const v8bf*)(aptr + k0 + akb + 8);
#pragma unroll
    for (int j = 0; j < 4; ++j) {
      v16bf b = *(const v16bf*)(cur + (j * 16 + lo) * BT_STRIDE + bkb);
      acc[j] = WMMA_BF16(a.v, b, acc[j]);
    }
  }
}

// ---------------------------------------------------------------------------
// Kernel: QKV projection.  C[4096 x 2304] = Xb[4096x768] * WqkvT + b_qkv
// Scatter into Q,K:[B,H,S,D]  V^T:[B,H,D,S] (bf16)
// ---------------------------------------------------------------------------
__global__ __launch_bounds__(256)
void qkv_gemm(const __bf16* __restrict__ xb, const __bf16* __restrict__ wt,
              const float* __restrict__ bias,
              __bf16* __restrict__ Qb, __bf16* __restrict__ Kb, __bf16* __restrict__ Vt) {
  __shared__ alignas(16) __bf16 btile[2 * BT_ELEMS];
  const int tid  = threadIdx.x;
  const int lane = tid & 31;
  const int wv   = tid >> 5;
  const int lo   = lane & 15;
  const int hi   = lane >> 4;
  const int m0   = blockIdx.y * 128 + wv * 16;
  const int n0   = blockIdx.x * 64;

  v8f acc[4] = {vzero8f(), vzero8f(), vzero8f(), vzero8f()};
  gemm_core(xb + (m0 + lo) * kE, wt, n0, tid, lo, hi, btile, acc);

#pragma unroll
  for (int j = 0; j < 4; ++j) {
    const int gcol  = n0 + j * 16 + lo;
    const int which = gcol / kE;           // 0=q 1=k 2=v (uniform per block)
    const int e     = gcol % kE;
    const int h     = e >> 6;
    const int d     = e & 63;
    const float bv  = bias[gcol];
#pragma unroll
    for (int i = 0; i < 8; ++i) {
      const int grow = m0 + i + 8 * hi;    // C layout: lane l, vgpr i -> row i+8*(l>>4)
      const int bi   = grow >> 11;
      const int s    = grow & (kS - 1);
      const float val = acc[j][i] + bv;
      const int bh = bi * kH + h;
      if (which == 0)      Qb[(bh * kS + s) * kD + d] = (__bf16)val;
      else if (which == 1) Kb[(bh * kS + s) * kD + d] = (__bf16)val;
      else                 Vt[(bh * kD + d) * kS + s] = (__bf16)val;
    }
  }
}

// ---------------------------------------------------------------------------
// Kernel: fused flash attention.  Per wave: 16 queries x full D=64.
// Online softmax over kv tiles of 32; P transposed C->A layout through LDS.
// ---------------------------------------------------------------------------
__global__ __launch_bounds__(256)
void attn_kernel(const __bf16* __restrict__ Qb, const __bf16* __restrict__ Kb,
                 const __bf16* __restrict__ Vt, __bf16* __restrict__ attnb) {
  __shared__ alignas(16) __bf16 plds[8][16 * 32];   // 8 KB, per-wave 16x32 P tile

  const int lane = threadIdx.x & 31;
  const int wv   = threadIdx.x >> 5;
  const int lo   = lane & 15;
  const int hi   = lane >> 4;
  const int bh   = blockIdx.x;                 // 0..23 = b*H+h
  const int bi   = bh / kH;
  const int h    = bh % kH;
  const int q0   = blockIdx.y * 128 + wv * 16;
  const int qkbase = bh * kS * kD;             // Q,K [B,H,S,D]
  const int vtbase = bh * kD * kS;             // V^T [B,H,D,S]

  // Q rows as two A-frags covering d = 0..31 and 32..63
  Afrag qa[2];
  const __bf16* qrow = Qb + qkbase + (q0 + lo) * kD;
#pragma unroll
  for (int c = 0; c < 2; ++c) {
    qa[c].h[0] = *(const v8bf*)(qrow + c * 32 + hi * 16);
    qa[c].h[1] = *(const v8bf*)(qrow + c * 32 + hi * 16 + 8);
  }

  v8f o[4] = {vzero8f(), vzero8f(), vzero8f(), vzero8f()};
  float m8[8], l8[8];
#pragma unroll
  for (int i = 0; i < 8; ++i) { m8[i] = -3.0e38f; l8[i] = 0.f; }
  const float scale = 0.125f;                  // 1/sqrt(64)

  __bf16* prow = &plds[wv][0];

  for (int kv0 = 0; kv0 < kS; kv0 += 32) {
    // --- scores S = Q * K^T : two 16x16 tiles (kv halves), K over d=64 ---
    v8f s0 = vzero8f(), s1 = vzero8f();
#pragma unroll
    for (int c = 0; c < 2; ++c) {
      v16bf bk0 = *(const v16bf*)(Kb + qkbase + (kv0 + lo) * kD      + c * 32 + hi * 16);
      v16bf bk1 = *(const v16bf*)(Kb + qkbase + (kv0 + 16 + lo) * kD + c * 32 + hi * 16);
      s0 = WMMA_BF16(qa[c].v, bk0, s0);
      s1 = WMMA_BF16(qa[c].v, bk1, s1);
    }

    // --- online softmax, per C-vgpr row (row m = i + 8*hi, spread over 16 lanes) ---
#pragma unroll
    for (int i = 0; i < 8; ++i) {
      float a0 = s0[i] * scale;
      float a1 = s1[i] * scale;
      float mx = fmaxf(a0, a1);
      mx = fmaxf(mx, __shfl_xor(mx, 1));
      mx = fmaxf(mx, __shfl_xor(mx, 2));
      mx = fmaxf(mx, __shfl_xor(mx, 4));
      mx = fmaxf(mx, __shfl_xor(mx, 8));
      const float mnew = fmaxf(m8[i], mx);
      const float corr = __expf(m8[i] - mnew);
      const float p0 = __expf(a0 - mnew);
      const float p1 = __expf(a1 - mnew);
      float ps = p0 + p1;
      ps += __shfl_xor(ps, 1);
      ps += __shfl_xor(ps, 2);
      ps += __shfl_xor(ps, 4);
      ps += __shfl_xor(ps, 8);
      l8[i] = l8[i] * corr + ps;
      m8[i] = mnew;
      o[0][i] *= corr; o[1][i] *= corr; o[2][i] *= corr; o[3][i] *= corr;
      const int m = i + 8 * hi;
      prow[m * 32 + lo]      = (__bf16)p0;     // P row-major 16x32 into LDS
      prow[m * 32 + 16 + lo] = (__bf16)p1;
    }

    // --- reload P in A-frag layout (LDS in-order within wave) ---
    Afrag pa;
    pa.h[0] = *(const v8bf*)(prow + lo * 32 + hi * 16);
    pa.h[1] = *(const v8bf*)(prow + lo * 32 + hi * 16 + 8);

    // --- O += P * V : 4 tiles over d ---
#pragma unroll
    for (int j = 0; j < 4; ++j) {
      v16bf bv = *(const v16bf*)(Vt + vtbase + (j * 16 + lo) * kS + kv0 + hi * 16);
      o[j] = WMMA_BF16(pa.v, bv, o[j]);
    }
  }

  // --- normalize + write attn (bf16, [B,S,E]) ---
#pragma unroll
  for (int i = 0; i < 8; ++i) {
    const float inv = 1.0f / l8[i];
    const int s = q0 + i + 8 * hi;
#pragma unroll
    for (int j = 0; j < 4; ++j) {
      const int e = h * 64 + j * 16 + lo;
      attnb[(bi * kS + s) * kE + e] = (__bf16)(o[j][i] * inv);
    }
  }
}

// ---------------------------------------------------------------------------
// Kernel: output projection.  out[4096 x 768] = Ab * WprojT + b_proj (fp32 out)
// ---------------------------------------------------------------------------
__global__ __launch_bounds__(256)
void proj_gemm(const __bf16* __restrict__ ab, const __bf16* __restrict__ wt,
               const float* __restrict__ bias, float* __restrict__ out) {
  __shared__ alignas(16) __bf16 btile[2 * BT_ELEMS];
  const int tid  = threadIdx.x;
  const int lane = tid & 31;
  const int wv   = tid >> 5;
  const int lo   = lane & 15;
  const int hi   = lane >> 4;
  const int m0   = blockIdx.y * 128 + wv * 16;
  const int n0   = blockIdx.x * 64;

  v8f acc[4] = {vzero8f(), vzero8f(), vzero8f(), vzero8f()};
  gemm_core(ab + (m0 + lo) * kE, wt, n0, tid, lo, hi, btile, acc);

#pragma unroll
  for (int j = 0; j < 4; ++j) {
    const int gcol = n0 + j * 16 + lo;
    const float bv = bias[gcol];
#pragma unroll
    for (int i = 0; i < 8; ++i) {
      const int grow = m0 + i + 8 * hi;
      out[grow * kE + gcol] = acc[j][i] + bv;
    }
  }
}

// ---------------------------------------------------------------------------
// Host launcher
// ---------------------------------------------------------------------------
extern "C" void kernel_launch(void* const* d_in, const int* in_sizes, int n_in,
                              void* d_out, int out_size, void* d_ws, size_t ws_size,
                              hipStream_t stream) {
  const float* x      = (const float*)d_in[0];
  const float* w_qkv  = (const float*)d_in[1];
  const float* b_qkv  = (const float*)d_in[2];
  const float* w_proj = (const float*)d_in[3];
  const float* b_proj = (const float*)d_in[4];
  float* out = (float*)d_out;

  char* ws = (char*)d_ws;
  size_t off = 0;
  auto alloc = [&](size_t bytes) -> void* {
    void* p = ws + off;
    off = (off + bytes + 255) & ~(size_t)255;
    return p;
  };
  __bf16* xb    = (__bf16*)alloc((size_t)kBS * kE * 2);        // x bf16
  __bf16* wqt   = (__bf16*)alloc((size_t)kE3 * kE * 2);        // w_qkv^T bf16
  __bf16* wpt   = (__bf16*)alloc((size_t)kE * kE * 2);         // w_proj^T bf16
  __bf16* Qb    = (__bf16*)alloc((size_t)kB * kH * kS * kD * 2);
  __bf16* Kb    = (__bf16*)alloc((size_t)kB * kH * kS * kD * 2);
  __bf16* Vt    = (__bf16*)alloc((size_t)kB * kH * kD * kS * 2);
  __bf16* attnb = (__bf16*)alloc((size_t)kBS * kE * 2);

  {
    int n4 = kBS * kE / 4;
    cvt_f32_bf16<<<(n4 + 255) / 256, 256, 0, stream>>>(x, xb, n4);
  }
  cvt_w_T<<<(kE * kE3 + 255) / 256, 256, 0, stream>>>(w_qkv, wqt, kE, kE3);
  cvt_w_T<<<(kE * kE + 255) / 256, 256, 0, stream>>>(w_proj, wpt, kE, kE);

  qkv_gemm<<<dim3(kE3 / 64, kBS / 128), 256, 0, stream>>>(xb, wqt, b_qkv, Qb, Kb, Vt);
  attn_kernel<<<dim3(kB * kH, kS / 128), 256, 0, stream>>>(Qb, Kb, Vt, attnb);
  proj_gemm<<<dim3(kE / 64, kBS / 128), 256, 0, stream>>>(attnb, wpt, b_proj, out);
}
